// BDHBlock_79877801771615
// MI455X (gfx1250) — matrix operand order repaired and checked by reading, not compiled, not run on hardware
//
#include <hip/hip_runtime.h>
#include <math.h>

#define DMODEL 768
#define NH     12
#define HD     64
#define BB     16
#define NN     1024
#define NTOK   (BB * NN)      /* 16384 */
#define EPSV   1e-5f
#define AW     8              /* waves per attention block */

typedef __bf16 bf16;
typedef __attribute__((ext_vector_type(16))) __bf16 v16bf;
typedef __attribute__((ext_vector_type(8)))  __bf16 bf16x8;
typedef __attribute__((ext_vector_type(8)))  float  v8f;
typedef __attribute__((ext_vector_type(4)))  int    i32x4;

// ---------------------------------------------------------------------------
// gfx1250 async global->LDS path (guarded; falls back to sync LDS copy)
// ---------------------------------------------------------------------------
#if defined(__AMDGCN__) && __has_builtin(__builtin_amdgcn_global_load_async_to_lds_b128)
#define HAS_ASYNC_LDS 1
#else
#define HAS_ASYNC_LDS 0
#endif

#if HAS_ASYNC_LDS
typedef __attribute__((address_space(1))) i32x4 as1_i32x4;
typedef __attribute__((address_space(3))) i32x4 as3_i32x4;
#endif

static __device__ __forceinline__ void async_cp16(const bf16* g, bf16* l) {
#if HAS_ASYNC_LDS
  __builtin_amdgcn_global_load_async_to_lds_b128(
      (as1_i32x4*)(void*)g, (as3_i32x4*)(void*)l, 0, 0);
#else
  *reinterpret_cast<bf16x8*>(l) = *reinterpret_cast<const bf16x8*>(g);
#endif
}

static __device__ __forceinline__ void wait_async_le2() {
#if HAS_ASYNC_LDS
#if __has_builtin(__builtin_amdgcn_s_wait_asynccnt)
  __builtin_amdgcn_s_wait_asynccnt(2);
#else
  asm volatile("s_wait_asynccnt 0x2" ::: "memory");
#endif
#endif
}

// ---------------------------------------------------------------------------
// WMMA helpers (gfx1250, wave32). D = A(16x32 bf16) * B(32x16 bf16) + C(f32)
// ---------------------------------------------------------------------------
static __device__ __forceinline__ v8f wmma_bf16(v16bf a, v16bf b, v8f c) {
  return __builtin_amdgcn_wmma_f32_16x16x32_bf16(
      /*neg_a=*/false, a, /*neg_b=*/false, b,
      /*c_mod=*/(short)0, c, /*reuse_a=*/false, /*reuse_b=*/false);
}

// A fragment 16x32: lane row = lane&15; lo-lanes hold K {0..7,16..23},
// hi-lanes K {8..15,24..31}  -> two contiguous 16B chunks per lane.
static __device__ __forceinline__ v16bf load_a(const bf16* base, int ld, int k0) {
  int lane = threadIdx.x & 31;
  const bf16* r = base + (size_t)(lane & 15) * ld + k0 + ((lane >> 4) << 3);
  bf16x8 lo = *reinterpret_cast<const bf16x8*>(r);
  bf16x8 hi = *reinterpret_cast<const bf16x8*>(r + 16);
  return __builtin_shufflevector(lo, hi, 0,1,2,3,4,5,6,7,8,9,10,11,12,13,14,15);
}

// B fragment 32x16 (KxN): column n = lane&15 is row (col0+n) of a row-major
// (Nout x K) source; lo-lanes K {0..7,16..23}, hi-lanes K {8..15,24..31}.
static __device__ __forceinline__ v16bf load_b(const bf16* base, int ld, int col0, int k0) {
  int lane = threadIdx.x & 31;
  const bf16* r = base + (size_t)(col0 + (lane & 15)) * ld + k0 + ((lane >> 4) << 3);
  bf16x8 lo = *reinterpret_cast<const bf16x8*>(r);
  bf16x8 hi = *reinterpret_cast<const bf16x8*>(r + 16);
  return __builtin_shufflevector(lo, hi, 0,1,2,3,4,5,6,7,8,9,10,11,12,13,14,15);
}

// ---------------------------------------------------------------------------
// Prep kernels
// ---------------------------------------------------------------------------
__global__ void k_cvt_bf16(const float* __restrict__ src, bf16* __restrict__ dst, int n) {
  int i = blockIdx.x * blockDim.x + threadIdx.x;
  if (i < n) dst[i] = (bf16)src[i];
}

// rope table per token: [cosA(32) | sinA(32) | sinB(32) | cosB(32)]
__global__ void k_rope_tab(const float* __restrict__ emb, float* __restrict__ tab) {
  int i = blockIdx.x * blockDim.x + threadIdx.x;   // NN*32 threads
  int n = i >> 5, j = i & 31;
  float a0 = emb[n * HD + 2 * j];
  float a1 = emb[n * HD + 2 * j + 1];
  float* t = tab + (size_t)n * 128;
  t[j]      = cosf(a0);
  t[32 + j] = sinf(a0);
  t[64 + j] = sinf(a1);
  t[96 + j] = cosf(a1);
}

// LayerNorm: one 256-thread block per row of 768, bf16 output
__global__ void k_ln(const float* __restrict__ x, const float* __restrict__ w,
                     const float* __restrict__ bia, bf16* __restrict__ xn) {
  __shared__ float s1[256], s2[256];
  int row = blockIdx.x, t = threadIdx.x;
  const float* xr = x + (size_t)row * DMODEL;
  float a = xr[t], b = xr[t + 256], c = xr[t + 512];
  s1[t] = a + b + c;
  s2[t] = a * a + b * b + c * c;
  __syncthreads();
  for (int off = 128; off; off >>= 1) {
    if (t < off) { s1[t] += s1[t + off]; s2[t] += s2[t + off]; }
    __syncthreads();
  }
  float mu  = s1[0] * (1.0f / DMODEL);
  float var = s2[0] * (1.0f / DMODEL) - mu * mu;
  float rs  = rsqrtf(var + EPSV);
  bf16* o = xn + (size_t)row * DMODEL;
  o[t]       = (bf16)((a - mu) * rs * w[t]       + bia[t]);
  o[t + 256] = (bf16)((b - mu) * rs * w[t + 256] + bia[t + 256]);
  o[t + 512] = (bf16)((c - mu) * rs * w[t + 512] + bia[t + 512]);
}

// ---------------------------------------------------------------------------
// GEMM: C(bf16) = relu(A(M,768) @ W(768,768)^T + bias). One wave -> 16x64 tile.
// ---------------------------------------------------------------------------
__global__ void k_gemm_relu(const bf16* __restrict__ A, const bf16* __restrict__ W,
                            const float* __restrict__ bias, bf16* __restrict__ C) {
  int wave = (blockIdx.x << 2) + (threadIdx.x >> 5);
  int ct = wave % (DMODEL / 64), rt = wave / (DMODEL / 64);
  int row0 = rt * 16, col0 = ct * 64;
  int lane = threadIdx.x & 31;
  const bf16* Ab = A + (size_t)row0 * DMODEL;
  v8f acc[4] = {};
  for (int k0 = 0; k0 < DMODEL; k0 += 32) {
    v16bf a = load_a(Ab, DMODEL, k0);
#pragma unroll
    for (int t = 0; t < 4; ++t)
      acc[t] = wmma_bf16(a, load_b(W, DMODEL, col0 + 16 * t, k0), acc[t]);
  }
  int col = lane & 15, rb = (lane >> 4) << 3;
#pragma unroll
  for (int t = 0; t < 4; ++t) {
    int cc = col0 + 16 * t + col;
    float bv = bias[cc];
    bf16* o = C + (size_t)(row0 + rb) * DMODEL + cc;
#pragma unroll
    for (int i = 0; i < 8; ++i) {
      float v = acc[t][i] + bv;
      o[(size_t)i * DMODEL] = (bf16)(v > 0.0f ? v : 0.0f);
    }
  }
}

// ---------------------------------------------------------------------------
// QK/V projection for one head x 16 tokens per wave, RoPE applied to qk.
// qk -> (B,H,N,64) row-major, v -> (B,H,64,N) transposed.
// ---------------------------------------------------------------------------
__global__ void k_qkv(const bf16* __restrict__ Xl, const bf16* __restrict__ Wqk,
                      const bf16* __restrict__ Wv, const float* __restrict__ bqk,
                      const float* __restrict__ bvv, const float* __restrict__ rtab,
                      bf16* __restrict__ QK, bf16* __restrict__ VT) {
  __shared__ float sQ[4][16][HD];
  int wid = threadIdx.x >> 5, lane = threadIdx.x & 31;
  int job = (blockIdx.x << 2) + wid;
  int h = job % NH, tile = job / NH;
  int row0 = tile * 16;
  int b = row0 / NN, n0 = row0 % NN;
  const bf16* Ab = Xl + (size_t)row0 * DMODEL;
  v8f qa[4] = {}, va[4] = {};
  for (int k0 = 0; k0 < DMODEL; k0 += 32) {
    v16bf a = load_a(Ab, DMODEL, k0);
#pragma unroll
    for (int t = 0; t < 4; ++t) {
      qa[t] = wmma_bf16(a, load_b(Wqk, DMODEL, h * HD + 16 * t, k0), qa[t]);
      va[t] = wmma_bf16(a, load_b(Wv,  DMODEL, h * HD + 16 * t, k0), va[t]);
    }
  }
  int col = lane & 15, rb = (lane >> 4) << 3;
  size_t bh = (size_t)b * NH + h;
  // V: bias + bf16 + transposed store: per lane 8 consecutive tokens (16B)
#pragma unroll
  for (int t = 0; t < 4; ++t) {
    int d = 16 * t + col;
    float bv = bvv[h * HD + d];
    bf16x8 pk;
#pragma unroll
    for (int i = 0; i < 8; ++i) pk[i] = (bf16)(va[t][i] + bv);
    *reinterpret_cast<bf16x8*>(VT + (bh * HD + d) * NN + n0 + rb) = pk;
  }
  // QK: bias, stage f32 tile in LDS, then RoPE re-layout + bf16 store
#pragma unroll
  for (int t = 0; t < 4; ++t) {
    int d = 16 * t + col;
    float bq = bqk[h * HD + d];
#pragma unroll
    for (int i = 0; i < 8; ++i) sQ[wid][rb + i][d] = qa[t][i] + bq;
  }
  __syncthreads();
  int tk = lane & 15, half = lane >> 4;
  const float* tb = rtab + (size_t)(n0 + tk) * 128 + half * 64;
  bf16* out = QK + (bh * NN + n0 + tk) * HD + half * 32;
#pragma unroll
  for (int c8 = 0; c8 < 4; ++c8) {
    bf16x8 pk;
#pragma unroll
    for (int u = 0; u < 8; ++u) {
      int j = c8 * 8 + u;
      float q1 = sQ[wid][tk][2 * j], q2 = sQ[wid][tk][2 * j + 1];
      float t0 = tb[j], t1 = tb[32 + j];
      float r = half ? (q1 * t0 + q2 * t1) : (q1 * t0 - q2 * t1);
      pk[u] = (bf16)r;
    }
    *reinterpret_cast<bf16x8*>(out + c8 * 8) = pk;
  }
}

// ---------------------------------------------------------------------------
// Attention (no softmax): A = (Q K^T / 8) V per (b,h).
// 8 waves x 16 rows; K/V tiles double-buffered in LDS via async global->LDS
// (ASYNCcnt), 8-way cross-wave reuse. S re-laid-out C->A through per-wave LDS.
// ---------------------------------------------------------------------------
__global__ void k_attn(const bf16* __restrict__ QK, const bf16* __restrict__ VT,
                       bf16* __restrict__ AO) {
  __shared__ __align__(32) bf16 sK[2][32][HD];   // K tile: 32 tokens x 64 dims
  __shared__ __align__(32) bf16 sV[2][HD][32];   // V^T tile: 64 dims x 32 tokens
  __shared__ __align__(32) bf16 sS[AW][16][32];  // per-wave S staging
  int tid = threadIdx.x, wid = tid >> 5, lane = tid & 31;
  int ntile = blockIdx.x % (NN / (16 * AW));
  int bh = blockIdx.x / (NN / (16 * AW));
  int b = bh / NH, h = bh % NH;
  int n0 = ntile * (16 * AW) + wid * 16;
  const bf16* Q  = QK + ((size_t)bh * NN + n0) * HD;
  const bf16* Kg = QK + (size_t)bh * NN * HD;
  const bf16* Vg = VT + (size_t)bh * HD * NN;
  v16bf a0 = load_a(Q, HD, 0);
  v16bf a1 = load_a(Q, HD, 32);
  v8f acc[4] = {};
  int col = lane & 15, rb = (lane >> 4) << 3;

  // stage K/V tile for m0 into buffer buf: 2 async ops (4KB each, 16B/thread)
  int kr = tid >> 3, kc = (tid & 7) * 8;   // K: 32 rows x 8 chunks
  int vd = tid >> 2, vc = (tid & 3) * 8;   // V: 64 rows x 4 chunks
#define STAGE(buf, m0)                                                     \
  do {                                                                     \
    async_cp16(Kg + (size_t)((m0) + kr) * HD + kc, &sK[buf][kr][kc]);      \
    async_cp16(Vg + (size_t)vd * NN + (m0) + vc, &sV[buf][vd][vc]);        \
  } while (0)

  STAGE(0, 0);
  for (int m0 = 0; m0 < NN; m0 += 32) {
    int buf = (m0 >> 5) & 1;
    if (m0 + 32 < NN) STAGE(buf ^ 1, m0 + 32);
    wait_async_le2();          // current buffer's copies complete
    __syncthreads();           // publish staged tiles to all waves
#pragma unroll
    for (int ctile = 0; ctile < 2; ++ctile) {
      v8f s = {};
      s = wmma_bf16(a0, load_b(&sK[buf][0][0], HD, 16 * ctile, 0),  s);
      s = wmma_bf16(a1, load_b(&sK[buf][0][0], HD, 16 * ctile, 32), s);
#pragma unroll
      for (int i = 0; i < 8; ++i)
        sS[wid][rb + i][16 * ctile + col] = (bf16)(s[i] * 0.125f);
    }
    v16bf as = load_a(&sS[wid][0][0], 32, 0);
#pragma unroll
    for (int t = 0; t < 4; ++t)
      acc[t] = wmma_bf16(as, load_b(&sV[buf][0][0], 32, 16 * t, 0), acc[t]);
    __syncthreads();           // all waves done with buf before re-staging it
  }
#undef STAGE
  // store to (NTOK, 768) rows, head h occupies cols [h*64, h*64+64)
#pragma unroll
  for (int t = 0; t < 4; ++t) {
    bf16* o = AO + ((size_t)(b * NN + n0 + rb)) * DMODEL + h * HD + 16 * t + col;
#pragma unroll
    for (int i = 0; i < 8; ++i) o[(size_t)i * DMODEL] = (bf16)acc[t][i];
  }
}

// ---------------------------------------------------------------------------
// Fused epilogue: y = x + sigmoid(xn@Wg^T+bg) * (attn@Wo^T+bo)
// ---------------------------------------------------------------------------
__global__ void k_epilogue(const bf16* __restrict__ AT, const bf16* __restrict__ XN,
                           const bf16* __restrict__ Wo, const bf16* __restrict__ Wg,
                           const float* __restrict__ bo, const float* __restrict__ bg,
                           const float* __restrict__ X, float* __restrict__ Y) {
  int wave = (blockIdx.x << 2) + (threadIdx.x >> 5);
  int ct = wave % (DMODEL / 64), rt = wave / (DMODEL / 64);
  int row0 = rt * 16, col0 = ct * 64;
  int lane = threadIdx.x & 31;
  const bf16* Aa = AT + (size_t)row0 * DMODEL;
  const bf16* Ax = XN + (size_t)row0 * DMODEL;
  v8f o[4] = {}, g[4] = {};
  for (int k0 = 0; k0 < DMODEL; k0 += 32) {
    v16bf aA = load_a(Aa, DMODEL, k0);
    v16bf aX = load_a(Ax, DMODEL, k0);
#pragma unroll
    for (int t = 0; t < 4; ++t) {
      o[t] = wmma_bf16(aA, load_b(Wo, DMODEL, col0 + 16 * t, k0), o[t]);
      g[t] = wmma_bf16(aX, load_b(Wg, DMODEL, col0 + 16 * t, k0), g[t]);
    }
  }
  int col = lane & 15, rb = (lane >> 4) << 3;
#pragma unroll
  for (int t = 0; t < 4; ++t) {
    int cc = col0 + 16 * t + col;
    float bov = bo[cc], bgv = bg[cc];
    const float* xi = X + (size_t)(row0 + rb) * DMODEL + cc;
    float* yo = Y + (size_t)(row0 + rb) * DMODEL + cc;
#pragma unroll
    for (int i = 0; i < 8; ++i) {
      float gate = 1.0f / (1.0f + expf(-(g[t][i] + bgv)));
      yo[(size_t)i * DMODEL] = xi[(size_t)i * DMODEL] + gate * (o[t][i] + bov);
    }
  }
}

// ---------------------------------------------------------------------------
extern "C" void kernel_launch(void* const* d_in, const int* in_sizes, int n_in,
                              void* d_out, int out_size, void* d_ws, size_t ws_size,
                              hipStream_t stream) {
  const float* x      = (const float*)d_in[0];
  const float* rope   = (const float*)d_in[1];
  const float* ln_w   = (const float*)d_in[2];
  const float* ln_b   = (const float*)d_in[3];
  const float* enc_w  = (const float*)d_in[4];
  const float* enc_b  = (const float*)d_in[5];
  const float* qk_w   = (const float*)d_in[6];
  const float* qk_b   = (const float*)d_in[7];
  const float* v_w    = (const float*)d_in[8];
  const float* v_b    = (const float*)d_in[9];
  const float* out_w  = (const float*)d_in[10];
  const float* out_b  = (const float*)d_in[11];
  const float* gate_w = (const float*)d_in[12];
  const float* gate_b = (const float*)d_in[13];
  float* y = (float*)d_out;

  char* p = (char*)d_ws;
  auto carve = [&](size_t bytes) -> char* {
    char* r = p; p += (bytes + 255) & ~(size_t)255; return r;
  };
  const size_t WSZ  = (size_t)DMODEL * DMODEL * sizeof(bf16);
  const size_t ACT  = (size_t)NTOK * DMODEL * sizeof(bf16);
  bf16*  encw  = (bf16*)carve(WSZ);
  bf16*  qkw   = (bf16*)carve(WSZ);
  bf16*  vw    = (bf16*)carve(WSZ);
  bf16*  outw  = (bf16*)carve(WSZ);
  bf16*  gatew = (bf16*)carve(WSZ);
  float* rtab  = (float*)carve((size_t)NN * 128 * sizeof(float));
  bf16*  xn    = (bf16*)carve(ACT);
  bf16*  xlat  = (bf16*)carve(ACT);
  bf16*  qkh   = (bf16*)carve(ACT);
  bf16*  vt    = (bf16*)carve(ACT);
  bf16*  attn  = xlat;  // x_latent is dead after k_qkv -> reuse for attn out

  const int wn = DMODEL * DMODEL;
  k_cvt_bf16<<<(wn + 255) / 256, 256, 0, stream>>>(enc_w,  encw,  wn);
  k_cvt_bf16<<<(wn + 255) / 256, 256, 0, stream>>>(qk_w,   qkw,   wn);
  k_cvt_bf16<<<(wn + 255) / 256, 256, 0, stream>>>(v_w,    vw,    wn);
  k_cvt_bf16<<<(wn + 255) / 256, 256, 0, stream>>>(out_w,  outw,  wn);
  k_cvt_bf16<<<(wn + 255) / 256, 256, 0, stream>>>(gate_w, gatew, wn);
  k_rope_tab<<<(NN * 32) / 256, 256, 0, stream>>>(rope, rtab);
  k_ln<<<NTOK, 256, 0, stream>>>(x, ln_w, ln_b, xn);

  const int gemm_blocks = (NTOK / 16) * (DMODEL / 64) / 4;   // 3072
  k_gemm_relu<<<gemm_blocks, 128, 0, stream>>>(xn, encw, enc_b, xlat);
  k_qkv<<<(NTOK / 16) * NH / 4, 128, 0, stream>>>(xlat, qkw, vw, qk_b, v_b,
                                                  rtab, qkh, vt);
  k_attn<<<BB * NH * (NN / (16 * AW)), 32 * AW, 0, stream>>>(qkh, vt, attn);
  k_epilogue<<<gemm_blocks, 128, 0, stream>>>(attn, xn, outw, gatew,
                                              out_b, gate_b, x, y);
}
